// GptOssMLP_86371792322903
// MI455X (gfx1250) — compile-verified, matrix operand order checked by reading, not run on hardware
//
#include <hip/hip_runtime.h>
#include <hip/hip_bf16.h>

// MoE MLP for MI455X (gfx1250, wave32, WMMA).
// Routed (top-4) grouped GEMMs in bf16 WMMA w/ fp32 accumulation.
// Double-buffered LDS pipeline, packed bf16 conversion.

#define NTOK 2048   // B*S
#define DDIM 1024
#define NEXP 8
#define IDIM 2048
#define TOPK 4

#define BM 128
#define BK 32
#define BN1 64      // GEMM1 column tile (gate & up computed together)
#define BN2 128     // GEMM2 column tile
#define LDT 40      // padded LDS row stride (halfs) to spread banks

typedef __bf16 v16bf __attribute__((ext_vector_type(16)));
typedef __bf16 v8bf  __attribute__((ext_vector_type(8)));
typedef float  v8f   __attribute__((ext_vector_type(8)));

__device__ __forceinline__ unsigned short f32_bf16(float f) {
  unsigned int u = __float_as_uint(f);
  u += 0x7fffu + ((u >> 16) & 1u);        // round-to-nearest-even
  return (unsigned short)(u >> 16);
}

#if __has_builtin(__builtin_amdgcn_cvt_pk_bf16_f32)
typedef __bf16 v2bf __attribute__((ext_vector_type(2)));
__device__ __forceinline__ unsigned int pk2(float a, float b) {
  v2bf r = __builtin_amdgcn_cvt_pk_bf16_f32(a, b);
  return __builtin_bit_cast(unsigned int, r);
}
#else
__device__ __forceinline__ unsigned int pk2(float a, float b) {
  return (unsigned int)f32_bf16(a) | ((unsigned int)f32_bf16(b) << 16);
}
#endif

__device__ __forceinline__ uint4 pk8(float4 a, float4 b) {
  return make_uint4(pk2(a.x, a.y), pk2(a.z, a.w), pk2(b.x, b.y), pk2(b.z, b.w));
}

// Load one 16x32 bf16 fragment from an LDS tile (row-major, stride ld halfs).
// Per CDNA5 ISA 16-bit A/B layout: lanes 0-15 hold rows 0-15 with K 0-7 /
// 16-23 in elems 0-7 / 8-15; lanes 16-31 hold K 8-15 / 24-31.
__device__ __forceinline__ v16bf frag_ld(const unsigned short* tile, int row0, int ld) {
  const int lane = threadIdx.x & 31;
  const int r  = row0 + (lane & 15);
  const int kb = (lane >> 4) * 8;
  const unsigned short* p = tile + r * ld + kb;
  v8bf lo = *(const v8bf*)(p);
  v8bf hi = *(const v8bf*)(p + 16);
  v16bf o;
#pragma unroll
  for (int i = 0; i < 8; ++i) { o[i] = lo[i]; o[i + 8] = hi[i]; }
  return o;
}

__device__ __forceinline__ v8f wmma_bf16(v16bf a, v16bf b, v8f c) {
  return __builtin_amdgcn_wmma_f32_16x16x32_bf16(
      /*neg_a=*/false, a, /*neg_b=*/false, b,
      /*c_mod=*/(short)0, c, /*reuse_a=*/false, /*reuse_b=*/false);
}

// ---------------------------------------------------------------- router ----
__global__ void __launch_bounds__(256)
router_kernel(const float* __restrict__ x, const float* __restrict__ rw,
              const float* __restrict__ rb, float* __restrict__ combine) {
  const int wave = threadIdx.x >> 5, lane = threadIdx.x & 31;
  const int n = blockIdx.x * 8 + wave;
  if (n >= NTOK) return;
  float s[NEXP];
#pragma unroll
  for (int e = 0; e < NEXP; ++e) s[e] = 0.f;
  const float* xp = x + (size_t)n * DDIM;
  for (int d = lane; d < DDIM; d += 32) {
    const float xv = xp[d];
#pragma unroll
    for (int e = 0; e < NEXP; ++e) s[e] += xv * rw[e * DDIM + d];
  }
#pragma unroll
  for (int e = 0; e < NEXP; ++e) {
    float v = s[e];
#pragma unroll
    for (int m = 16; m >= 1; m >>= 1) v += __shfl_xor(v, m, 32);
    s[e] = v + rb[e];
  }
  // top-4 with lowest-index tie-break (matches jax.lax.top_k), then softmax
  bool used[NEXP];
#pragma unroll
  for (int e = 0; e < NEXP; ++e) used[e] = false;
  float tv[TOPK]; int ti[TOPK];
#pragma unroll
  for (int k = 0; k < TOPK; ++k) {
    float best = -3.4e38f; int bi = 0;
#pragma unroll
    for (int e = 0; e < NEXP; ++e)
      if (!used[e] && s[e] > best) { best = s[e]; bi = e; }
    used[bi] = true; tv[k] = best; ti[k] = bi;
  }
  float p[TOPK], ssum = 0.f;
#pragma unroll
  for (int k = 0; k < TOPK; ++k) { p[k] = __expf(tv[k] - tv[0]); ssum += p[k]; }
  float w[NEXP];
#pragma unroll
  for (int e = 0; e < NEXP; ++e) w[e] = 0.f;
#pragma unroll
  for (int k = 0; k < TOPK; ++k) w[ti[k]] = p[k] / ssum;
  if (lane < NEXP) combine[n * NEXP + lane] = w[lane];
}

// ------------------------------------------------------- per-expert scan ----
__global__ void __launch_bounds__(256)
scan_kernel(const float* __restrict__ combine, int* __restrict__ tokenlist,
            int* __restrict__ row_of, int* __restrict__ counts) {
  const int e = blockIdx.x;
  const int tid = threadIdx.x;
  __shared__ int sbuf[256];
  __shared__ int srun;
  if (tid == 0) srun = 0;
  __syncthreads();
  for (int base = 0; base < NTOK; base += 256) {
    const int n = base + tid;
    const int flag = (combine[n * NEXP + e] > 0.f) ? 1 : 0;
    int v = flag;
    sbuf[tid] = v;
    __syncthreads();
#pragma unroll
    for (int off = 1; off < 256; off <<= 1) {
      const int add = (tid >= off) ? sbuf[tid - off] : 0;
      __syncthreads();
      v += add;
      sbuf[tid] = v;
      __syncthreads();
    }
    const int run = srun;
    const int total = sbuf[255];
    if (flag) {
      const int pos = run + v - 1;
      tokenlist[e * NTOK + pos] = n;
      row_of[n * NEXP + e] = pos;
    }
    __syncthreads();
    if (tid == 0) srun = run + total;
    __syncthreads();
  }
  if (tid == 0) counts[e] = srun;
}

// ------------------------------------------- GEMM1: x @ gate_up^T + swiglu ---
// C[m][j] = sum_d x[tok_m][d] * gate_up[e][j][d]; gate (j) and up (j+IDIM)
// tiles computed together, fused clamp/sigmoid, bf16 act out.
__global__ void __launch_bounds__(256)
gemm1_swiglu(const float* __restrict__ x, const float* __restrict__ gup,
             const float* __restrict__ gub, const int* __restrict__ tokenlist,
             const int* __restrict__ counts, unsigned short* __restrict__ act) {
  const int e = blockIdx.z;
  const int count = counts[e];
  const int tm = blockIdx.y, tn = blockIdx.x;
  if (tm * BM >= count) return;

  __shared__ unsigned short As[2][BM * LDT];
  __shared__ unsigned short Bg[2][BN1 * LDT];
  __shared__ unsigned short Bu[2][BN1 * LDT];

  const int t = threadIdx.x;
  // A loader: 16 floats/thread from gathered token row, convert to bf16
  const int ar = t >> 1, ac = (t & 1) * 16;
  const int mg_a = tm * BM + ar;
  const int tok = (mg_a < count) ? tokenlist[e * NTOK + mg_a] : -1;
  const float* xrow = x + (size_t)(tok < 0 ? 0 : tok) * DDIM;
  // B loaders: 8 floats/thread from weight rows (K contiguous)
  const int br = t >> 2, bc = (t & 3) * 8;
  const int jg = tn * BN1 + br;
  const float* gsrc = gup + ((size_t)e * 2 * IDIM + jg) * DDIM;
  const float* usrc = gup + ((size_t)e * 2 * IDIM + IDIM + jg) * DDIM;

  const int wv = t >> 5, lane = t & 31;
  const int wm = wv & 3, wn = wv >> 2;   // 4x2 wave grid, 32x32 per wave

  v8f accg[2][2], accu[2][2];
#pragma unroll
  for (int i = 0; i < 2; ++i)
#pragma unroll
    for (int j = 0; j < 2; ++j) { accg[i][j] = (v8f)(0.f); accu[i][j] = (v8f)(0.f); }

  uint4 ra0, ra1, rg, ru;
  auto g2r = [&](int k0) {
    if (tok >= 0) {
      const float4 f0 = *(const float4*)(xrow + k0 + ac);
      const float4 f1 = *(const float4*)(xrow + k0 + ac + 4);
      const float4 f2 = *(const float4*)(xrow + k0 + ac + 8);
      const float4 f3 = *(const float4*)(xrow + k0 + ac + 12);
      ra0 = pk8(f0, f1);
      ra1 = pk8(f2, f3);
    } else {
      ra0 = make_uint4(0, 0, 0, 0);
      ra1 = make_uint4(0, 0, 0, 0);
    }
    {
      const float4 g0 = *(const float4*)(gsrc + k0 + bc);
      const float4 g1 = *(const float4*)(gsrc + k0 + bc + 4);
      rg = pk8(g0, g1);
    }
    {
      const float4 u0 = *(const float4*)(usrc + k0 + bc);
      const float4 u1 = *(const float4*)(usrc + k0 + bc + 4);
      ru = pk8(u0, u1);
    }
    if (k0 + 2 * BK <= DDIM - BK) {
      __builtin_prefetch(gsrc + k0 + 2 * BK + bc, 0, 0);
      __builtin_prefetch(usrc + k0 + 2 * BK + bc, 0, 0);
    }
  };

  g2r(0);
  for (int k0 = 0; k0 < DDIM; k0 += BK) {
    const int st = (k0 >> 5) & 1;
    *(uint4*)&As[st][ar * LDT + ac]     = ra0;
    *(uint4*)&As[st][ar * LDT + ac + 8] = ra1;
    *(uint4*)&Bg[st][br * LDT + bc]     = rg;
    *(uint4*)&Bu[st][br * LDT + bc]     = ru;
    if (k0 + BK < DDIM) g2r(k0 + BK);       // overlap next tile's loads
    __syncthreads();

    v16bf af[2], bgf[2], buf[2];
    af[0]  = frag_ld(As[st], wm * 32, LDT);  af[1]  = frag_ld(As[st], wm * 32 + 16, LDT);
    bgf[0] = frag_ld(Bg[st], wn * 32, LDT);  bgf[1] = frag_ld(Bg[st], wn * 32 + 16, LDT);
    buf[0] = frag_ld(Bu[st], wn * 32, LDT);  buf[1] = frag_ld(Bu[st], wn * 32 + 16, LDT);
#pragma unroll
    for (int i = 0; i < 2; ++i)
#pragma unroll
      for (int j = 0; j < 2; ++j) {
        accg[i][j] = wmma_bf16(af[i], bgf[j], accg[i][j]);
        accu[i][j] = wmma_bf16(af[i], buf[j], accu[i][j]);
      }
  }

  // fused clamped-SwiGLU epilogue -> bf16 act
  const int ln = lane & 15, lh = lane >> 4;
#pragma unroll
  for (int i = 0; i < 2; ++i)
#pragma unroll
    for (int j = 0; j < 2; ++j) {
      const int jcol = tn * BN1 + wn * 32 + j * 16 + ln;
      const float gb = gub[e * 2 * IDIM + jcol];
      const float ub = gub[e * 2 * IDIM + IDIM + jcol];
#pragma unroll
      for (int v = 0; v < 8; ++v) {
        const int m = tm * BM + wm * 32 + i * 16 + lh * 8 + v;
        if (m < count) {
          float g = accg[i][j][v] + gb;
          float u = accu[i][j][v] + ub;
          g = fminf(g, 7.f);
          u = fminf(fmaxf(u, -7.f), 7.f);
          const float sig = 1.f / (1.f + __expf(-1.702f * g));
          const float a = (u + 1.f) * (g * sig);
          act[((size_t)e * NTOK + m) * IDIM + jcol] = f32_bf16(a);
        }
      }
    }
}

// ----------------------------------------------- GEMM2: act @ down^T + b ----
__global__ void __launch_bounds__(256)
gemm2_kernel(const unsigned short* __restrict__ act, const float* __restrict__ dwn,
             const float* __restrict__ dbias, const int* __restrict__ counts,
             float* __restrict__ outbuf) {
  const int e = blockIdx.z;
  const int count = counts[e];
  const int tm = blockIdx.y, tn = blockIdx.x;
  if (tm * BM >= count) return;

  __shared__ unsigned short As[2][BM * LDT];
  __shared__ unsigned short Bs[2][BN2 * LDT];

  const int t = threadIdx.x;
  const int ar = t >> 1, ac = (t & 1) * 16;    // A: 16 halfs/thread (bf16 src)
  const int mg_a = tm * BM + ar;
  const unsigned short* asrc = act + ((size_t)e * NTOK + mg_a) * IDIM;
  const int br = t >> 1, bc = (t & 1) * 16;    // B: 16 floats/thread
  const int dg = tn * BN2 + br;
  const float* bsrc = dwn + ((size_t)e * DDIM + dg) * IDIM;

  const int wv = t >> 5, lane = t & 31;
  const int wm = wv & 3, wn = wv >> 2;   // 4x2 wave grid, 32x64 per wave

  v8f acc[2][4];
#pragma unroll
  for (int i = 0; i < 2; ++i)
#pragma unroll
    for (int j = 0; j < 4; ++j) acc[i][j] = (v8f)(0.f);

  uint4 ra0, ra1, rb0, rb1;
  auto g2r = [&](int k0) {
    ra0 = *(const uint4*)(asrc + k0 + ac);
    ra1 = *(const uint4*)(asrc + k0 + ac + 8);
    const float4 f0 = *(const float4*)(bsrc + k0 + bc);
    const float4 f1 = *(const float4*)(bsrc + k0 + bc + 4);
    const float4 f2 = *(const float4*)(bsrc + k0 + bc + 8);
    const float4 f3 = *(const float4*)(bsrc + k0 + bc + 12);
    rb0 = pk8(f0, f1);
    rb1 = pk8(f2, f3);
    if (k0 + 2 * BK <= IDIM - BK)
      __builtin_prefetch(bsrc + k0 + 2 * BK + bc, 0, 0);
  };

  g2r(0);
  for (int k0 = 0; k0 < IDIM; k0 += BK) {
    const int st = (k0 >> 5) & 1;
    *(uint4*)&As[st][ar * LDT + ac]     = ra0;
    *(uint4*)&As[st][ar * LDT + ac + 8] = ra1;
    *(uint4*)&Bs[st][br * LDT + bc]     = rb0;
    *(uint4*)&Bs[st][br * LDT + bc + 8] = rb1;
    if (k0 + BK < IDIM) g2r(k0 + BK);
    __syncthreads();

    v16bf af[2], bf[4];
    af[0] = frag_ld(As[st], wm * 32, LDT);
    af[1] = frag_ld(As[st], wm * 32 + 16, LDT);
#pragma unroll
    for (int j = 0; j < 4; ++j) bf[j] = frag_ld(Bs[st], wn * 64 + j * 16, LDT);
#pragma unroll
    for (int i = 0; i < 2; ++i)
#pragma unroll
      for (int j = 0; j < 4; ++j)
        acc[i][j] = wmma_bf16(af[i], bf[j], acc[i][j]);
  }

  const int ln = lane & 15, lh = lane >> 4;
#pragma unroll
  for (int i = 0; i < 2; ++i)
#pragma unroll
    for (int j = 0; j < 4; ++j) {
      const int dcol = tn * BN2 + wn * 64 + j * 16 + ln;
      const float db = dbias[e * DDIM + dcol];
#pragma unroll
      for (int v = 0; v < 8; ++v) {
        const int m = tm * BM + wm * 32 + i * 16 + lh * 8 + v;
        if (m < count)
          outbuf[((size_t)e * NTOK + m) * DDIM + dcol] = acc[i][j][v] + db;
      }
    }
}

// ------------------------------------------------------- weighted combine ---
__global__ void __launch_bounds__(256)
combine_kernel(const float* __restrict__ combine, const int* __restrict__ row_of,
               const float* __restrict__ outbuf, float* __restrict__ y) {
  const int n = blockIdx.x;
  const int d = threadIdx.x * 4;
  float4 a = make_float4(0.f, 0.f, 0.f, 0.f);
#pragma unroll
  for (int e = 0; e < NEXP; ++e) {          // fixed order -> deterministic
    const float c = combine[n * NEXP + e];
    if (c > 0.f) {
      const int r = row_of[n * NEXP + e];
      const float4 o = *(const float4*)&outbuf[((size_t)e * NTOK + r) * DDIM + d];
      a.x += c * o.x; a.y += c * o.y; a.z += c * o.z; a.w += c * o.w;
    }
  }
  *(float4*)&y[(size_t)n * DDIM + d] = a;
}

// --------------------------------------------------------------- launcher ---
extern "C" void kernel_launch(void* const* d_in, const int* in_sizes, int n_in,
                              void* d_out, int out_size, void* d_ws, size_t ws_size,
                              hipStream_t stream) {
  const float* x     = (const float*)d_in[0];  // [N, D]
  const float* rw    = (const float*)d_in[1];  // [E, D]
  const float* rb    = (const float*)d_in[2];  // [E]
  const float* gup   = (const float*)d_in[3];  // [E, 2I, D]
  const float* gub   = (const float*)d_in[4];  // [E, 2I]
  const float* dwn   = (const float*)d_in[5];  // [E, D, I]
  const float* dbias = (const float*)d_in[6];  // [E, D]
  float* y = (float*)d_out;                    // [N, D]

  char* w = (char*)d_ws;
  float* combine = (float*)w;           w += (size_t)NTOK * NEXP * 4;
  int*   row_of  = (int*)w;             w += (size_t)NTOK * NEXP * 4;
  int*   tlist   = (int*)w;             w += (size_t)NEXP * NTOK * 4;
  int*   counts  = (int*)w;             w += 256;
  unsigned short* act = (unsigned short*)w; w += (size_t)NEXP * NTOK * IDIM * 2;
  float* outbuf  = (float*)w;           // NEXP*NTOK*DDIM*4 bytes

  router_kernel<<<NTOK / 8, 256, 0, stream>>>(x, rw, rb, combine);
  scan_kernel<<<NEXP, 256, 0, stream>>>(combine, tlist, row_of, counts);

  dim3 g1(IDIM / BN1, NTOK / BM, NEXP);   // (32, 16, 8)
  gemm1_swiglu<<<g1, 256, 0, stream>>>(x, gup, gub, tlist, counts, act);

  dim3 g2(DDIM / BN2, NTOK / BM, NEXP);   // (8, 16, 8)
  gemm2_kernel<<<g2, 256, 0, stream>>>(act, dwn, dbias, counts, outbuf);

  combine_kernel<<<NTOK, 256, 0, stream>>>(combine, row_of, outbuf, y);
}
